// VGAEGraphEncoder_67516885893468
// MI455X (gfx1250) — matrix operand order repaired and checked by reading, not compiled, not run on hardware
//
#include <hip/hip_runtime.h>
#include <hip/hip_bf16.h>

typedef __attribute__((ext_vector_type(16))) __bf16 v16bf;
typedef __attribute__((ext_vector_type(8)))  float  v8f;

#define HIDF 128

__device__ __forceinline__ unsigned short f32_to_bf16_bits(float f) {
    unsigned u = __float_as_uint(f);
    unsigned r = u + 0x7FFFu + ((u >> 16) & 1u);   // round-to-nearest-even
    return (unsigned short)(r >> 16);
}

__device__ __forceinline__ void atomic_add_f32(float* p, float v) {
    __hip_atomic_fetch_add(p, v, __ATOMIC_RELAXED, __HIP_MEMORY_SCOPE_AGENT);
}

// ---------------------------------------------------------------------------
// Zero-fill
// ---------------------------------------------------------------------------
__global__ void zero_f32(float* __restrict__ p, long long n) {
    long long i = (long long)blockIdx.x * blockDim.x + threadIdx.x;
    if (i < n) p[i] = 0.0f;
}

// ---------------------------------------------------------------------------
// Degree accumulation: deg[dst[e]] += 1
// ---------------------------------------------------------------------------
__global__ void deg_kernel(const int* __restrict__ dst, float* __restrict__ deg, int E) {
    int e = blockIdx.x * blockDim.x + threadIdx.x;
    if (e < E) atomic_add_f32(&deg[dst[e]], 1.0f);
}

// dinv = rsqrt(deg + 1)  (in place)
__global__ void dinv_kernel(float* __restrict__ deg, int n) {
    int i = blockIdx.x * blockDim.x + threadIdx.x;
    if (i < n) deg[i] = __frsqrt_rn(deg[i] + 1.0f);
}

// ---------------------------------------------------------------------------
// GEMM: Y[N x 128] = X[N x 128] @ W[128 x 128] using bf16 WMMA, f32 acc.
// Block = 256 threads = 8 waves; each block does a 128-row slab; each wave
// a 16-row stripe across all 128 output cols (8 tiles x 4 k-steps = 32 WMMA).
// ---------------------------------------------------------------------------
__global__ __launch_bounds__(256)
void gemm128_bf16_wmma(const float* __restrict__ X, const float* __restrict__ W,
                       float* __restrict__ Y, int n_rows) {
    __shared__ unsigned short lds_wT[HIDF * HIDF]; // W^T bf16: [n][k], k contiguous
    __shared__ unsigned short lds_x [HIDF * HIDF]; // X slab bf16: [rowLocal][k]

    const int tid  = threadIdx.x;
    const int lane = tid & 31;
    const int wave = tid >> 5;              // 0..7
    const int row0 = blockIdx.x * 128;

    // Stage W (global row-major [k][n]) transposed into LDS as bf16.
    for (int i = tid; i < HIDF * HIDF; i += 256) {
        int k = i >> 7, n = i & 127;
        lds_wT[n * HIDF + k] = f32_to_bf16_bits(W[i]);
    }
    // Stage a 128-row slab of X as bf16 (zero-pad past n_rows).
    for (int i = tid; i < HIDF * HIDF; i += 256) {
        int r = i >> 7, k = i & 127;
        int gr = row0 + r;
        float v = (gr < n_rows) ? X[(long long)gr * HIDF + k] : 0.0f;
        lds_x[i] = f32_to_bf16_bits(v);
    }
    __syncthreads();

    const int mrow   = lane & 15;
    const int aKhalf = (lane >> 4) * 16;    // A: lane half -> K offset 0 / 16
    const int bKhalf = (lane >> 4) * 16;    // B: lane half -> K offset 0 / 16
    const int ncol   = lane & 15;

    v8f zero = {};
    v8f acc[8];
#pragma unroll
    for (int n = 0; n < 8; ++n) acc[n] = zero;

    const unsigned short* xrow = &lds_x[(wave * 16 + mrow) * HIDF];

#pragma unroll
    for (int ks = 0; ks < 4; ++ks) {
        const int kbase = ks * 32;
        // A fragment: 8 contiguous bf16 pairs per lane (contiguous k = 16*half + s, same order as B)
        union { unsigned u[8]; v16bf v; } afrag;
#pragma unroll
        for (int r = 0; r < 8; ++r) {
            int K = kbase + aKhalf + r * 2;
            afrag.u[r] = *(const unsigned*)(&xrow[K]);
        }
#pragma unroll
        for (int n = 0; n < 8; ++n) {
            union { unsigned u[8]; v16bf v; } bfrag;
            const unsigned short* wrow =
                &lds_wT[(n * 16 + ncol) * HIDF + kbase + bKhalf];
#pragma unroll
            for (int r = 0; r < 8; ++r)
                bfrag.u[r] = *(const unsigned*)(&wrow[r * 2]);
            acc[n] = __builtin_amdgcn_wmma_f32_16x16x32_bf16(
                false, afrag.v, false, bfrag.v, (short)0, acc[n], false, false);
        }
    }

    // Store: C layout — VGPR r holds row r + 8*(lane>=16), col = lane&15.
    const int rbase = wave * 16 + (lane >> 4) * 8;
#pragma unroll
    for (int n = 0; n < 8; ++n) {
        int col = n * 16 + ncol;
#pragma unroll
        for (int r = 0; r < 8; ++r) {
            int grow = row0 + rbase + r;
            if (grow < n_rows) Y[(long long)grow * HIDF + col] = acc[n][r];
        }
    }
}

// ---------------------------------------------------------------------------
// Edge scatter: one wave per edge; 32 lanes x float4 covers 128 features.
// Agg[dst] += coef * H[src], coef = dinv[src]*dinv[dst]
// ---------------------------------------------------------------------------
__global__ __launch_bounds__(256)
void edge_scatter(const float* __restrict__ H, const int* __restrict__ src,
                  const int* __restrict__ dst, const float* __restrict__ dinv,
                  float* __restrict__ Agg, int E) {
    int e = blockIdx.x * 8 + (threadIdx.x >> 5);
    if (e >= E) return;
    int lane = threadIdx.x & 31;
    int s = src[e], d = dst[e];
    float c = dinv[s] * dinv[d];
    const float4 hv = ((const float4*)(H + (long long)s * HIDF))[lane];
    float* a = Agg + (long long)d * HIDF + lane * 4;
    atomic_add_f32(a + 0, c * hv.x);
    atomic_add_f32(a + 1, c * hv.y);
    atomic_add_f32(a + 2, c * hv.z);
    atomic_add_f32(a + 3, c * hv.w);
}

// ---------------------------------------------------------------------------
// Epilogue: Out = Agg + self_coef * HW + bias, optional ELU. In-place on Agg.
// ---------------------------------------------------------------------------
template <bool DO_ELU>
__global__ void self_bias_kernel(float* __restrict__ Agg, const float* __restrict__ HW,
                                 const float* __restrict__ dinv,
                                 const float* __restrict__ bias, long long total) {
    long long i = (long long)blockIdx.x * blockDim.x + threadIdx.x;
    if (i >= total) return;
    int node = (int)(i >> 7);
    int f    = (int)(i & 127);
    float di = dinv[node];
    float v  = Agg[i] + di * di * HW[i] + bias[f];
    if (DO_ELU) v = (v > 0.0f) ? v : (__expf(v) - 1.0f);
    Agg[i] = v;
}

// ---------------------------------------------------------------------------
// Reparameterize: z = mu + eps * exp(0.5*logvar), eps ~ N(0,1) via hash.
// (JAX PRNG stream not reproducible on-device; deterministic hash normal.)
// ---------------------------------------------------------------------------
__device__ __forceinline__ unsigned hash_u32(unsigned x) {
    x ^= x >> 17; x *= 0xed5ad4bbu;
    x ^= x >> 11; x *= 0xac4c1b51u;
    x ^= x >> 15; x *= 0x31848babu;
    x ^= x >> 14; return x;
}

__global__ void reparam_kernel(const float* __restrict__ mu, const float* __restrict__ lv,
                               float* __restrict__ z, long long total) {
    long long i = (long long)blockIdx.x * blockDim.x + threadIdx.x;
    if (i >= total) return;
    unsigned h1 = hash_u32((unsigned)i * 2u + 0x9E3779B9u);
    unsigned h2 = hash_u32((unsigned)i * 2u + 0x85EBCA77u);
    float u1 = (float)(h1 >> 8) * (1.0f / 16777216.0f) + 1.1754944e-38f;
    float u2 = (float)(h2 >> 8) * (1.0f / 16777216.0f);
    float eps = __fsqrt_rn(-2.0f * __logf(u1)) * __cosf(6.28318530718f * u2);
    z[i] = mu[i] + eps * __expf(0.5f * lv[i]);
}

// ---------------------------------------------------------------------------
// Launch
// ---------------------------------------------------------------------------
extern "C" void kernel_launch(void* const* d_in, const int* in_sizes, int n_in,
                              void* d_out, int out_size, void* d_ws, size_t ws_size,
                              hipStream_t stream) {
    const float* x    = (const float*)d_in[0];
    const int*   ei   = (const int*)d_in[1];
    const float* W1   = (const float*)d_in[2];
    const float* b1   = (const float*)d_in[3];
    const float* Wmu  = (const float*)d_in[4];
    const float* bmu  = (const float*)d_in[5];
    const float* Wlv  = (const float*)d_in[6];
    const float* blv  = (const float*)d_in[7];

    const int N = in_sizes[0] / HIDF;
    const int E = in_sizes[1] / 2;
    const int* src = ei;
    const int* dst = ei + E;

    const long long NF = (long long)N * HIDF;

    // Workspace: dinv[N] | hw[N*128] | h[N*128]
    float* dinv = (float*)d_ws;
    float* hw   = dinv + N;
    float* h    = hw + NF;

    float* out    = (float*)d_out;
    float* z_sec  = out;
    float* mu_sec = out + NF;
    float* lv_sec = out + 2 * NF;

    const int T = 256;
    dim3 blkE((E + 7) / 8);          // 8 edges (waves) per 256-thread block
    dim3 blkG((N + 127) / 128);      // 128-row slabs
    dim3 blkNF((unsigned)((NF + T - 1) / T));
    dim3 blkN((N + T - 1) / T);
    dim3 blkEth((E + T - 1) / T);

    // ---- norm coefficients ----
    zero_f32<<<blkN, T, 0, stream>>>(dinv, N);
    deg_kernel<<<blkEth, T, 0, stream>>>(dst, dinv, E);
    dinv_kernel<<<blkN, T, 0, stream>>>(dinv, N);

    // ---- layer 1: h = elu(gcn(x, W1, b1)) ----
    gemm128_bf16_wmma<<<blkG, T, 0, stream>>>(x, W1, hw, N);
    zero_f32<<<blkNF, T, 0, stream>>>(h, NF);
    edge_scatter<<<blkE, T, 0, stream>>>(hw, src, dst, dinv, h, E);
    self_bias_kernel<true><<<blkNF, T, 0, stream>>>(h, hw, dinv, b1, NF);

    // ---- mu = gcn(h, W_mu, b_mu) ----
    gemm128_bf16_wmma<<<blkG, T, 0, stream>>>(h, Wmu, hw, N);
    zero_f32<<<blkNF, T, 0, stream>>>(mu_sec, NF);
    edge_scatter<<<blkE, T, 0, stream>>>(hw, src, dst, dinv, mu_sec, E);
    self_bias_kernel<false><<<blkNF, T, 0, stream>>>(mu_sec, hw, dinv, bmu, NF);

    // ---- logvar = gcn(h, W_lv, b_lv) ----
    gemm128_bf16_wmma<<<blkG, T, 0, stream>>>(h, Wlv, hw, N);
    zero_f32<<<blkNF, T, 0, stream>>>(lv_sec, NF);
    edge_scatter<<<blkE, T, 0, stream>>>(hw, src, dst, dinv, lv_sec, E);
    self_bias_kernel<false><<<blkNF, T, 0, stream>>>(lv_sec, hw, dinv, blv, NF);

    // ---- z = mu + eps * exp(0.5*logvar) ----
    reparam_kernel<<<blkNF, T, 0, stream>>>(mu_sec, lv_sec, z_sec, NF);
}
